// MolecularE3nnQm9_7164005449943
// MI455X (gfx1250) — compile-verified, matrix operand order checked by reading, not run on hardware
//
#include <hip/hip_runtime.h>
#include <hip/hip_bf16.h>
#include <math.h>

// ---------------------------------------------------------------------------
// MolecularE3nnQm9 forward for MI455X (gfx1250, wave32, WMMA).
// All dense GEMMs (radial MLP r1/r2, per-edge emb@tp_w, z-grouped node _lin's)
// run on v_wmma_f32_16x16x32_f16 (f16 in, f32 accumulate). Equivariant
// message construction + CSR segment reduction run on the VALU in f32.
// ---------------------------------------------------------------------------

#define NATOMS 6000
#define NGRAPH 300
#define NTMAX  ((NATOMS / 16) + 11)       // max z-grouped 16-row tiles (nodes)
#define NT3MAX ((NATOMS * 3 / 16) + 11)   // max tiles for node-component rows

#define SQ3C  1.7320508075688772f
#define SQ5C  2.2360679774997896f
#define SQ15C 3.8729833462074170f
#define A3C   0.5773502691896258f    // 1/sqrt(3)
#define S6I   0.4082482904638630f    // 1/sqrt(6)
#define K1C   0.3162277660168379f    // 1/sqrt(10)  (C121 off-diag entries)
#define K2C   0.1825741858350554f    // 1/sqrt(30)  (C121 b=2 entries)
#define INV272 0.0606339062590832f   // 1/sqrt(272)
#define INVS20 0.2236067977499790f   // 1/sqrt(20)

typedef __attribute__((ext_vector_type(16))) _Float16 v16h;
typedef __attribute__((ext_vector_type(8)))  float    v8f;

union HFrag { v16h v; uint4 q[2]; };
static_assert(sizeof(HFrag) == 32, "frag size");

__device__ __forceinline__ v8f wmma_f16(const HFrag& a, const HFrag& b, v8f c) {
  return __builtin_amdgcn_wmma_f32_16x16x32_f16(false, a.v, false, b.v,
                                                (short)0, c, false, false);
}

// -------------------------- weight conversion ------------------------------
// W (U,10,Wd) f32 -> Bt (10, Wd, Kpad) f16 (column-major B per z, zero K-pad)
__global__ void w_to_bt(const float* __restrict__ W, _Float16* __restrict__ Bt,
                        int U, int Wd, int Kpad) {
  int i = blockIdx.x * blockDim.x + threadIdx.x;
  int total = 10 * Wd * Kpad;
  if (i >= total) return;
  int k = i % Kpad;
  int w = (i / Kpad) % Wd;
  int z = i / (Kpad * Wd);
  Bt[i] = (k < U) ? (_Float16)W[(size_t)k * 10 * Wd + (size_t)z * Wd + w]
                  : (_Float16)0.f;
}

// W (K,Ncols) f32 -> Bt (Ncols, Kpad) f16
__global__ void m_to_bt(const float* __restrict__ W, _Float16* __restrict__ Bt,
                        int K, int Ncols, int Kpad) {
  int i = blockIdx.x * blockDim.x + threadIdx.x;
  int total = Ncols * Kpad;
  if (i >= total) return;
  int k = i % Kpad;
  int n = i / Kpad;
  Bt[i] = (k < K) ? (_Float16)W[(size_t)k * Ncols + n] : (_Float16)0.f;
}

// -------------------------- edge GEMM (WMMA) -------------------------------
// C(M x N, f16) = epilogue(scale * A(M x K, f16, row-major) @ B) with B given
// as Bt (N x Kpad, f16).  One wave per 16x16 tile.  EPI: 0 = identity, 1 = silu.
template <int EPI>
__global__ void gemm_edge(const _Float16* __restrict__ A, int lda,
                          const _Float16* __restrict__ Bt, int ldb,
                          _Float16* __restrict__ C, int ldc,
                          int M, int ksteps, float scale) {
  int m0 = blockIdx.x * 16, n0 = blockIdx.y * 16;
  int lane = threadIdx.x;
  int arow = m0 + (lane & 15);
  int brow = n0 + (lane & 15);
  int koff = (lane >> 4) ? 8 : 0;
  const _Float16* ap = A + (size_t)arow * lda + koff;
  const _Float16* bp = Bt + (size_t)brow * ldb + koff;
  v8f acc = {};
#pragma unroll 4
  for (int ks = 0; ks < ksteps; ++ks) {
    int k0 = ks * 32;
    HFrag a, b;
    a.q[0] = *reinterpret_cast<const uint4*>(ap + k0);
    a.q[1] = *reinterpret_cast<const uint4*>(ap + k0 + 16);
    b.q[0] = *reinterpret_cast<const uint4*>(bp + k0);
    b.q[1] = *reinterpret_cast<const uint4*>(bp + k0 + 16);
    acc = wmma_f16(a, b, acc);
  }
  int nc = n0 + (lane & 15);
  int mbase = m0 + ((lane >> 4) << 3);
  for (int r = 0; r < 8; ++r) {
    int m = mbase + r;
    if (m < M) {
      float v = acc[r] * scale;
      if (EPI == 1) v = v / (1.f + expf(-v));  // silu
      C[(size_t)m * ldc + nc] = (_Float16)v;
    }
  }
}

// ------------------ z-grouped node linear (WMMA, gather/scatter) -----------
// Rows outside the tile's valid count read from a zeroed dummy row so the
// fragment loads stay unconditional (no EXEC-mask branching in the K loop).
__global__ void lin_grouped(const _Float16* __restrict__ A, int lda,
                            const _Float16* __restrict__ Bt, long long zstride,
                            int ldb, float* __restrict__ Out, int ldo,
                            const int* __restrict__ perm,
                            const int* __restrict__ tz,
                            const int* __restrict__ ts,
                            const int* __restrict__ tc,
                            const int* __restrict__ ntile,
                            const _Float16* __restrict__ zrow,
                            int ksteps, float scale) {
  int t = blockIdx.x;
  if (t >= *ntile) return;
  int cnt = tc[t];
  int z = tz[t];
  int tstart = ts[t];
  int lane = threadIdx.x;
  __shared__ int rows[16];
  if (lane < 16) rows[lane] = (lane < cnt) ? perm[tstart + lane] : -1;
  __syncthreads();
  int my = rows[lane & 15];
  int n0 = blockIdx.y * 16;
  const _Float16* Bz = Bt + (size_t)z * zstride;
  int brow = n0 + (lane & 15);
  int koff = (lane >> 4) ? 8 : 0;
  const _Float16* ap = (my >= 0) ? (A + (size_t)my * lda + koff) : (zrow + koff);
  const _Float16* bp = Bz + (size_t)brow * ldb + koff;
  v8f acc = {};
#pragma unroll 3
  for (int ks = 0; ks < ksteps; ++ks) {
    int k0 = ks * 32;
    HFrag a, b;
    a.q[0] = *reinterpret_cast<const uint4*>(ap + k0);
    a.q[1] = *reinterpret_cast<const uint4*>(ap + k0 + 16);
    b.q[0] = *reinterpret_cast<const uint4*>(bp + k0);
    b.q[1] = *reinterpret_cast<const uint4*>(bp + k0 + 16);
    acc = wmma_f16(a, b, acc);
  }
  int nc = n0 + (lane & 15);
  int ib = (lane >> 4) << 3;
  for (int r = 0; r < 8; ++r) {
    int row = rows[ib + r];
    if (row >= 0) Out[(size_t)row * ldo + nc] = acc[r] * scale;
  }
}

// --------------------------- graph/CSR building ----------------------------
__global__ void count_deg(const int* __restrict__ dst, int E, int* deg) {
  int e = blockIdx.x * blockDim.x + threadIdx.x;
  if (e < E) atomicAdd(&deg[dst[e]], 1);
}
__global__ void count_z(const int* __restrict__ z, int* zcnt) {
  int n = blockIdx.x * blockDim.x + threadIdx.x;
  if (n < NATOMS) atomicAdd(&zcnt[z[n]], 1);
}
__global__ void build_tables(const int* __restrict__ deg, int* rowptr,
                             int* fillpos, const int* __restrict__ zcnt,
                             int* zstart, int* zfill, int* tz, int* ts, int* tc,
                             int* nt, int* tz3, int* ts3, int* tc3, int* nt3) {
  if (threadIdx.x != 0 || blockIdx.x != 0) return;
  int acc = 0;
  for (int n = 0; n < NATOMS; ++n) {
    rowptr[n] = acc;
    fillpos[n] = acc;
    acc += deg[n];
  }
  rowptr[NATOMS] = acc;
  int zs = 0, t = 0, t3 = 0;
  for (int z = 0; z < 10; ++z) {
    zstart[z] = zs;
    zfill[z] = zs;
    int c = zcnt[z];
    for (int off = 0; off < c; off += 16) {
      tz[t] = z; ts[t] = zs + off; tc[t] = (c - off < 16) ? (c - off) : 16; ++t;
    }
    int c3 = 3 * c, zs3 = 3 * zs;
    for (int off = 0; off < c3; off += 16) {
      tz3[t3] = z; ts3[t3] = zs3 + off;
      tc3[t3] = (c3 - off < 16) ? (c3 - off) : 16; ++t3;
    }
    zs += c;
  }
  zstart[10] = zs;
  *nt = t;
  *nt3 = t3;
}
__global__ void fill_eidx(const int* __restrict__ dst, int E, int* fillpos,
                          int* eidx) {
  int e = blockIdx.x * blockDim.x + threadIdx.x;
  if (e < E) { int p = atomicAdd(&fillpos[dst[e]], 1); eidx[p] = e; }
}
__global__ void fill_perm(const int* __restrict__ z, int* zfill, int* perm) {
  int n = blockIdx.x * blockDim.x + threadIdx.x;
  if (n < NATOMS) { int p = atomicAdd(&zfill[z[n]], 1); perm[p] = n; }
}
__global__ void build_perm3(const int* __restrict__ perm, int* perm3) {
  int j = blockIdx.x * blockDim.x + threadIdx.x;
  if (j < NATOMS) {
    int node = perm[j];
    perm3[3 * j + 0] = node * 3 + 0;
    perm3[3 * j + 1] = node * 3 + 1;
    perm3[3 * j + 2] = node * 3 + 2;
  }
}

// ----------------------- edge geometry + radial basis ----------------------
__global__ void edge_geom(const float* __restrict__ pos,
                          const int* __restrict__ src,
                          const int* __restrict__ dst, int E,
                          float* __restrict__ eY0, float* __restrict__ eY1,
                          float* __restrict__ eY2, _Float16* __restrict__ soh) {
  int e = blockIdx.x * blockDim.x + threadIdx.x;
  if (e >= E) return;
  int a = src[e], b = dst[e];
  float vx = pos[a * 3 + 0] - pos[b * 3 + 0];
  float vy = pos[a * 3 + 1] - pos[b * 3 + 1];
  float vz = pos[a * 3 + 2] - pos[b * 3 + 2];
  float r = sqrtf(vx * vx + vy * vy + vz * vz);
  float inv = 1.f / r;
  float x = vx * inv, y = vy * inv, z = vz * inv;
  float c = 0.5f * (cosf(0.31415926535f * r) + 1.f) * INVS20;
  eY0[e] = c;
  eY1[e * 3 + 0] = c * SQ3C * x;
  eY1[e * 3 + 1] = c * SQ3C * y;
  eY1[e * 3 + 2] = c * SQ3C * z;
  eY2[e * 5 + 0] = c * SQ15C * x * y;
  eY2[e * 5 + 1] = c * SQ15C * y * z;
  eY2[e * 5 + 2] = c * (SQ5C * 0.5f) * (2.f * z * z - x * x - y * y);
  eY2[e * 5 + 3] = c * SQ15C * x * z;
  eY2[e * 5 + 4] = c * (SQ15C * 0.5f) * (x * x - y * y);
  const float step = 10.f / 51.f;
  for (int g = 0; g < 64; ++g) {
    float f = 0.f;
    if (g < 50) {
      float d = (r - (float)(g + 1) * step) / step;
      float p = d + 1.f, q = 1.f - d;
      float sp = p > 0.f ? expf(-1.f / fmaxf(p, 1e-6f)) : 0.f;
      float sq = q > 0.f ? expf(-1.f / fmaxf(q, 1e-6f)) : 0.f;
      f = 8.4335731f * sp * sq;  // 1.14136 * e^2
    }
    soh[(size_t)e * 64 + g] = (_Float16)f;
  }
}

__global__ void h1_accum(const int* __restrict__ esrc,
                         const float* __restrict__ eY1, float* h1, int E) {
  int e = blockIdx.x * blockDim.x + threadIdx.x;
  if (e >= E) return;
  int s = esrc[e];
  atomicAdd(&h1[s * 3 + 0], eY1[e * 3 + 0]);
  atomicAdd(&h1[s * 3 + 1], eY1[e * 3 + 1]);
  atomicAdd(&h1[s * 3 + 2], eY1[e * 3 + 2]);
}

__global__ void init_nodes(const int* __restrict__ z,
                           const float* __restrict__ mn_s,
                           const float* __restrict__ mn_vo,
                           const float* __restrict__ h1,
                           _Float16* __restrict__ s_h,
                           _Float16* __restrict__ vo_h,
                           _Float16* __restrict__ ve_h) {
  int n = blockIdx.x, t = threadIdx.x, zz = z[n];
  s_h[(size_t)n * 256 + t] = (_Float16)mn_s[zz * 256 + t];
  if (t < 96) {
    int m = t >> 5, u = t & 31;
    float v = (u < 16) ? h1[n * 3 + m] * mn_vo[zz * 16 + u] * A3C : 0.f;
    vo_h[((size_t)n * 3 + m) * 32 + u] = (_Float16)v;
    ve_h[((size_t)n * 3 + m) * 32 + u] = (_Float16)0.f;
  }
}

// --------------------- conv message build + segment sum --------------------
__global__ void conv_aggregate(const int* __restrict__ rowptr,
                               const int* __restrict__ eidx,
                               const int* __restrict__ esrc,
                               const float* __restrict__ eY0,
                               const float* __restrict__ eY1,
                               const float* __restrict__ eY2,
                               const _Float16* __restrict__ wb,
                               const float* __restrict__ s1,
                               const float* __restrict__ vo1,
                               const float* __restrict__ ve1,
                               _Float16* __restrict__ n_s_h,
                               _Float16* __restrict__ n_vo_h,
                               _Float16* __restrict__ n_ve_h) {
  int n = blockIdx.x, t = threadIdx.x;  // 256 threads
  __shared__ float accs[288];
  __shared__ float accvo[3][288];
  __shared__ float accve[3][16];
  for (int i = t; i < 288; i += 256) {
    accs[i] = 0.f; accvo[0][i] = 0.f; accvo[1][i] = 0.f; accvo[2][i] = 0.f;
  }
  if (t < 48) accve[t / 16][t & 15] = 0.f;
  __syncthreads();
  int e0 = rowptr[n], e1 = rowptr[n + 1];
  for (int ei = e0; ei < e1; ++ei) {
    int e = eidx[ei];
    if (ei + 1 < e1)
      __builtin_prefetch(wb + (size_t)eidx[ei + 1] * 624, 0, 0);
    int s = esrc[e];
    float Y0 = eY0[e];
    float y10 = eY1[e * 3], y11 = eY1[e * 3 + 1], y12 = eY1[e * 3 + 2];
    const _Float16* wrow = wb + (size_t)e * 624;
    {  // 256-channel scalar part: every thread owns slot t (no conflicts)
      float w00 = (float)wrow[t], w01 = (float)wrow[256 + t];
      float xs = s1[(size_t)s * 256 + t];
      accs[t] += w00 * xs * Y0;
      float bb = w01 * xs * A3C;
      accvo[0][t] += bb * y10;
      accvo[1][t] += bb * y11;
      accvo[2][t] += bb * y12;
    }
    if (t < 16) {  // 16-channel equivariant part: thread t owns slot 256+t
      int u = t;
      float y20 = eY2[e * 5], y21 = eY2[e * 5 + 1], y22 = eY2[e * 5 + 2];
      float y23 = eY2[e * 5 + 3], y24 = eY2[e * 5 + 4];
      const float* vob = vo1 + (size_t)s * 48;
      const float* veb = ve1 + (size_t)s * 48;
      float xo0 = vob[u], xo1 = vob[16 + u], xo2 = vob[32 + u];
      float xe0 = veb[u], xe1 = veb[16 + u], xe2 = veb[32 + u];
      float wr0 = (float)wrow[512 + u], wr1 = (float)wrow[528 + u];
      float wr2 = (float)wrow[544 + u], wr3 = (float)wrow[560 + u];
      float wr4 = (float)wrow[576 + u], wr5 = (float)wrow[592 + u];
      float wr6 = (float)wrow[608 + u];
      float dot_o = xo0 * y10 + xo1 * y11 + xo2 * y12;
      accs[256 + u] += wr1 * dot_o * A3C;
      // cg121(xo, eY2)   (C121 entries: +-1/sqrt(10), +-1/sqrt(30))
      float co0 = K1C * (xo1 * y20 + xo2 * y23 + xo0 * y24) - K2C * xo0 * y22;
      float co1 = K1C * (xo0 * y20 + xo2 * y21 - xo1 * y24) - K2C * xo1 * y22;
      float co2 = K1C * (xo1 * y21 + xo0 * y23) + 2.f * K2C * xo2 * y22;
      // cg121(xe, eY2)
      float ce0 = K1C * (xe1 * y20 + xe2 * y23 + xe0 * y24) - K2C * xe0 * y22;
      float ce1 = K1C * (xe0 * y20 + xe2 * y21 - xe1 * y24) - K2C * xe1 * y22;
      float ce2 = K1C * (xe1 * y21 + xe0 * y23) + 2.f * K2C * xe2 * y22;
      // cg111 = cross / sqrt(6)
      float he0 = (xe1 * y12 - xe2 * y11) * S6I;
      float he1 = (xe2 * y10 - xe0 * y12) * S6I;
      float he2 = (xe0 * y11 - xe1 * y10) * S6I;
      float ho0 = (xo1 * y12 - xo2 * y11) * S6I;
      float ho1 = (xo2 * y10 - xo0 * y12) * S6I;
      float ho2 = (xo0 * y11 - xo1 * y10) * S6I;
      float a3y0 = A3C * Y0;
      accvo[0][256 + u] += A3C * (wr0 * xo0 * a3y0 + wr3 * co0 + wr5 * he0);
      accvo[1][256 + u] += A3C * (wr0 * xo1 * a3y0 + wr3 * co1 + wr5 * he1);
      accvo[2][256 + u] += A3C * (wr0 * xo2 * a3y0 + wr3 * co2 + wr5 * he2);
      accve[0][u] += A3C * (wr2 * ho0 + wr4 * xe0 * a3y0 + wr6 * ce0);
      accve[1][u] += A3C * (wr2 * ho1 + wr4 * xe1 * a3y0 + wr6 * ce1);
      accve[2][u] += A3C * (wr2 * ho2 + wr4 * xe2 * a3y0 + wr6 * ce2);
    }
  }
  __syncthreads();
  for (int i = t; i < 288; i += 256) {
    n_s_h[(size_t)n * 288 + i] = (_Float16)(i < 272 ? accs[i] : 0.f);
    for (int m = 0; m < 3; ++m)
      n_vo_h[((size_t)n * 3 + m) * 288 + i] =
          (_Float16)(i < 272 ? accvo[m][i] : 0.f);
  }
  if (t < 96) {
    int m = t >> 5, u = t & 31;
    n_ve_h[((size_t)n * 3 + m) * 32 + u] =
        (_Float16)(u < 16 ? accve[m][u] : 0.f);
  }
}

// ----------------------------- gating / update -----------------------------
__global__ void update_nodes(const float* __restrict__ s_si,
                             const float* __restrict__ l2s,
                             const float* __restrict__ vo_si,
                             const float* __restrict__ l2vo,
                             const float* __restrict__ ve_si,
                             const float* __restrict__ l2ve,
                             _Float16* __restrict__ s_h,
                             _Float16* __restrict__ vo_h,
                             _Float16* __restrict__ ve_h) {
  int n = blockIdx.x, t = threadIdx.x;  // 288 threads
  __shared__ float g[32];
  float pre = s_si[(size_t)n * 288 + t] + 0.1f * l2s[(size_t)n * 288 + t];
  if (t >= 256) g[t - 256] = 1.f / (1.f + expf(-pre));
  __syncthreads();
  if (t < 256) s_h[(size_t)n * 256 + t] = (_Float16)tanhf(pre);
  if (t < 96) {
    int m = t >> 5, u = t & 31;
    if (u < 16) {
      size_t r = ((size_t)n * 3 + m) * 16 + u;
      float vv = (vo_si[r] + 0.1f * l2vo[r]) * g[u];
      float ev = (ve_si[r] + 0.1f * l2ve[r]) * g[16 + u];
      vo_h[((size_t)n * 3 + m) * 32 + u] = (_Float16)vv;
      ve_h[((size_t)n * 3 + m) * 32 + u] = (_Float16)ev;
    } else {
      vo_h[((size_t)n * 3 + m) * 32 + u] = (_Float16)0.f;
      ve_h[((size_t)n * 3 + m) * 32 + u] = (_Float16)0.f;
    }
  }
}

// ------------------------------ output head --------------------------------
__global__ void fin_aggregate(const int* __restrict__ rowptr,
                              const int* __restrict__ eidx,
                              const int* __restrict__ esrc,
                              const float* __restrict__ eY0,
                              const float* __restrict__ eY1,
                              const _Float16* __restrict__ wf,
                              const float* __restrict__ s1,
                              const float* __restrict__ vo1,
                              const float* __restrict__ ve1,
                              const _Float16* __restrict__ s_h,
                              const int* __restrict__ zarr,
                              const float* __restrict__ Wsis,
                              const float* __restrict__ Wl2s,
                              const float* __restrict__ Wl2o,
                              float* __restrict__ hnode) {
  int n = blockIdx.x, t = threadIdx.x;  // 288 threads
  __shared__ float acce[272];
  __shared__ float acco[16];
  __shared__ float red[288];
  __shared__ float redo[288];
  if (t < 272) acce[t] = 0.f;
  if (t < 16) acco[t] = 0.f;
  __syncthreads();
  int e0 = rowptr[n], e1 = rowptr[n + 1];
  for (int ei = e0; ei < e1; ++ei) {
    int e = eidx[ei];
    int s = esrc[e];
    float Y0 = eY0[e];
    const _Float16* wrow = wf + (size_t)e * 288;
    if (t < 256) {
      float xs = s1[(size_t)s * 256 + t];
      acce[t] += (float)wrow[t] * xs * Y0;
    }
    if (t < 16) {
      float y10 = eY1[e * 3], y11 = eY1[e * 3 + 1], y12 = eY1[e * 3 + 2];
      const float* vob = vo1 + (size_t)s * 48;
      const float* veb = ve1 + (size_t)s * 48;
      float dot_o = vob[t] * y10 + vob[16 + t] * y11 + vob[32 + t] * y12;
      float dot_e = veb[t] * y10 + veb[16 + t] * y11 + veb[32 + t] * y12;
      acce[256 + t] += (float)wrow[256 + t] * dot_o * A3C;
      acco[t] += (float)wrow[272 + t] * dot_e * A3C;
    }
  }
  __syncthreads();
  int z = zarr[n];
  float p = 0.f;
  if (t < 256) p += (float)s_h[(size_t)n * 256 + t] * Wsis[t * 10 + z] * 0.0625f;
  if (t < 272) p += acce[t] * Wl2s[t * 10 + z] * INV272 * 0.1f;
  red[t] = p;
  redo[t] = (t < 16) ? acco[t] * Wl2o[t * 10 + z] * 0.25f * 0.1f : 0.f;
  __syncthreads();
  if (t == 0) {
    float se = 0.f, so = 0.f;
    for (int i = 0; i < 288; ++i) { se += red[i]; so += redo[i]; }
    hnode[n] = (se + 0.5f * so * so) * INVS20;
  }
}

__global__ void graph_out(const float* __restrict__ hnode,
                          const int* __restrict__ batch, float* out) {
  int n = blockIdx.x * blockDim.x + threadIdx.x;
  if (n < NATOMS) atomicAdd(&out[batch[n]], hnode[n]);
}

// ===========================================================================
extern "C" void kernel_launch(void* const* d_in, const int* in_sizes, int n_in,
                              void* d_out, int out_size, void* d_ws,
                              size_t ws_size, hipStream_t stream) {
  (void)n_in; (void)ws_size; (void)out_size;
  // Input order = setup_inputs() insertion order:
  // z, pos, edge_src, edge_dst, batch, mn_s, mn_vo, r1, r2,
  // 3x layer {si_s, si_vo, si_ve, l1_s, l1_vo, l1_ve, tp_w, l2_s, l2_vo, l2_ve},
  // fin {si_s, l1_s, l1_vo, l1_ve, tp_w, l2_s, l2_o}
  const int*   z     = (const int*)d_in[0];
  const float* pos   = (const float*)d_in[1];
  const int*   esrc  = (const int*)d_in[2];
  const int*   edst  = (const int*)d_in[3];
  const int*   batch = (const int*)d_in[4];
  const float* mn_s  = (const float*)d_in[5];
  const float* mn_vo = (const float*)d_in[6];
  const float* r1w   = (const float*)d_in[7];
  const float* r2w   = (const float*)d_in[8];
  const float *Lsi_s[3], *Lsi_vo[3], *Lsi_ve[3], *Ll1_s[3], *Ll1_vo[3],
      *Ll1_ve[3], *Ltp[3], *Ll2_s[3], *Ll2_vo[3], *Ll2_ve[3];
  for (int i = 0; i < 3; ++i) {
    int b = 9 + 10 * i;
    Lsi_s[i]  = (const float*)d_in[b + 0];
    Lsi_vo[i] = (const float*)d_in[b + 1];
    Lsi_ve[i] = (const float*)d_in[b + 2];
    Ll1_s[i]  = (const float*)d_in[b + 3];
    Ll1_vo[i] = (const float*)d_in[b + 4];
    Ll1_ve[i] = (const float*)d_in[b + 5];
    Ltp[i]    = (const float*)d_in[b + 6];
    Ll2_s[i]  = (const float*)d_in[b + 7];
    Ll2_vo[i] = (const float*)d_in[b + 8];
    Ll2_ve[i] = (const float*)d_in[b + 9];
  }
  const float* Fsi_s  = (const float*)d_in[39];
  const float* Fl1_s  = (const float*)d_in[40];
  const float* Fl1_vo = (const float*)d_in[41];
  const float* Fl1_ve = (const float*)d_in[42];
  const float* Ftp    = (const float*)d_in[43];
  const float* Fl2_s  = (const float*)d_in[44];
  const float* Fl2_o  = (const float*)d_in[45];

  const int E = in_sizes[2];
  const int Epad = (E + 15) & ~15;

  char* cur = (char*)d_ws;
  auto take = [&](size_t bytes) -> char* {
    char* p = cur;
    cur += (bytes + 255) & ~(size_t)255;
    return p;
  };
  // ---- integer scratch
  int* deg     = (int*)take(NATOMS * 4);
  int* rowptr  = (int*)take((NATOMS + 1) * 4);
  int* fillpos = (int*)take(NATOMS * 4);
  int* eidx    = (int*)take((size_t)E * 4);
  int* zcnt    = (int*)take(16 * 4);
  int* zstart  = (int*)take(16 * 4);
  int* zfill   = (int*)take(16 * 4);
  int* perm    = (int*)take(NATOMS * 4);
  int* perm3   = (int*)take((size_t)NATOMS * 3 * 4);
  int* tz  = (int*)take(NTMAX * 4);
  int* ts  = (int*)take(NTMAX * 4);
  int* tc  = (int*)take(NTMAX * 4);
  int* tz3 = (int*)take(NT3MAX * 4);
  int* ts3 = (int*)take(NT3MAX * 4);
  int* tc3 = (int*)take(NT3MAX * 4);
  int* ntile  = (int*)take(4);
  int* ntile3 = (int*)take(4);
  // ---- f32 scratch
  float* eY0   = (float*)take((size_t)E * 4);
  float* eY1   = (float*)take((size_t)E * 12);
  float* eY2   = (float*)take((size_t)E * 20);
  float* h1    = (float*)take(NATOMS * 12);
  float* s1    = (float*)take((size_t)NATOMS * 256 * 4);
  float* vo1   = (float*)take((size_t)NATOMS * 48 * 4);
  float* ve1   = (float*)take((size_t)NATOMS * 48 * 4);
  float* s_si  = (float*)take((size_t)NATOMS * 288 * 4);
  float* vo_si = (float*)take((size_t)NATOMS * 48 * 4);
  float* ve_si = (float*)take((size_t)NATOMS * 48 * 4);
  float* l2s   = (float*)take((size_t)NATOMS * 288 * 4);
  float* l2vo  = (float*)take((size_t)NATOMS * 48 * 4);
  float* l2ve  = (float*)take((size_t)NATOMS * 48 * 4);
  float* hnode = (float*)take(NATOMS * 4);
  // ---- f16 activations
  _Float16* soh   = (_Float16*)take((size_t)Epad * 64 * 2);
  _Float16* emb1  = (_Float16*)take((size_t)Epad * 128 * 2);
  _Float16* emb   = (_Float16*)take((size_t)Epad * 128 * 2);
  _Float16* wbuf  = (_Float16*)take((size_t)Epad * 624 * 2);
  _Float16* s_h   = (_Float16*)take((size_t)NATOMS * 256 * 2);
  _Float16* vo_h  = (_Float16*)take((size_t)NATOMS * 96 * 2);
  _Float16* ve_h  = (_Float16*)take((size_t)NATOMS * 96 * 2);
  _Float16* ns_h  = (_Float16*)take((size_t)NATOMS * 288 * 2);
  _Float16* nvo_h = (_Float16*)take((size_t)NATOMS * 864 * 2);
  _Float16* nve_h = (_Float16*)take((size_t)NATOMS * 96 * 2);
  _Float16* zrow  = (_Float16*)take(320 * 2);   // zeroed dummy A-row
  // ---- f16 weight (Bt) buffers
  _Float16 *bt_sis[3], *bt_l1s[3], *bt_l2s[3], *bt_l2vo[3], *bt_svo[3],
      *bt_sve[3], *bt_l1vo[3], *bt_l1ve[3], *bt_l2ve[3], *bt_tp[3];
  for (int i = 0; i < 3; ++i) {
    bt_sis[i]  = (_Float16*)take((size_t)10 * 288 * 256 * 2);
    bt_l1s[i]  = (_Float16*)take((size_t)10 * 256 * 256 * 2);
    bt_l2s[i]  = (_Float16*)take((size_t)10 * 288 * 288 * 2);
    bt_l2vo[i] = (_Float16*)take((size_t)10 * 16 * 288 * 2);
    bt_svo[i]  = (_Float16*)take((size_t)10 * 16 * 32 * 2);
    bt_sve[i]  = (_Float16*)take((size_t)10 * 16 * 32 * 2);
    bt_l1vo[i] = (_Float16*)take((size_t)10 * 16 * 32 * 2);
    bt_l1ve[i] = (_Float16*)take((size_t)10 * 16 * 32 * 2);
    bt_l2ve[i] = (_Float16*)take((size_t)10 * 16 * 32 * 2);
    bt_tp[i]   = (_Float16*)take((size_t)624 * 128 * 2);
  }
  _Float16* bt_fl1s  = (_Float16*)take((size_t)10 * 256 * 256 * 2);
  _Float16* bt_fl1vo = (_Float16*)take((size_t)10 * 16 * 32 * 2);
  _Float16* bt_fl1ve = (_Float16*)take((size_t)10 * 16 * 32 * 2);
  _Float16* bt_ftp   = (_Float16*)take((size_t)288 * 128 * 2);
  _Float16* bt_r1    = (_Float16*)take((size_t)128 * 64 * 2);
  _Float16* bt_r2    = (_Float16*)take((size_t)128 * 128 * 2);

  auto conv_lin = [&](const float* W, _Float16* B, int U, int Wd, int Kp) {
    int tot = 10 * Wd * Kp;
    w_to_bt<<<(tot + 255) / 256, 256, 0, stream>>>(W, B, U, Wd, Kp);
  };
  auto conv_mat = [&](const float* W, _Float16* B, int K, int Nc, int Kp) {
    int tot = Nc * Kp;
    m_to_bt<<<(tot + 255) / 256, 256, 0, stream>>>(W, B, K, Nc, Kp);
  };

  // ---- weight prep (idempotent, every call)
  conv_mat(r1w, bt_r1, 50, 128, 64);
  conv_mat(r2w, bt_r2, 128, 128, 128);
  for (int i = 0; i < 3; ++i) {
    conv_lin(Lsi_s[i], bt_sis[i], 256, 288, 256);
    conv_lin(Ll1_s[i], bt_l1s[i], 256, 256, 256);
    conv_lin(Ll2_s[i], bt_l2s[i], 272, 288, 288);
    conv_lin(Ll2_vo[i], bt_l2vo[i], 272, 16, 288);
    conv_lin(Lsi_vo[i], bt_svo[i], 16, 16, 32);
    conv_lin(Lsi_ve[i], bt_sve[i], 16, 16, 32);
    conv_lin(Ll1_vo[i], bt_l1vo[i], 16, 16, 32);
    conv_lin(Ll1_ve[i], bt_l1ve[i], 16, 16, 32);
    conv_lin(Ll2_ve[i], bt_l2ve[i], 16, 16, 32);
    conv_mat(Ltp[i], bt_tp[i], 128, 624, 128);
  }
  conv_lin(Fl1_s, bt_fl1s, 256, 256, 256);
  conv_lin(Fl1_vo, bt_fl1vo, 16, 16, 32);
  conv_lin(Fl1_ve, bt_fl1ve, 16, 16, 32);
  conv_mat(Ftp, bt_ftp, 128, 288, 128);

  // ---- CSR (by dst) + z-grouping
  hipMemsetAsync(deg, 0, NATOMS * 4, stream);
  hipMemsetAsync(zcnt, 0, 16 * 4, stream);
  hipMemsetAsync(zrow, 0, 320 * 2, stream);
  count_deg<<<(E + 255) / 256, 256, 0, stream>>>(edst, E, deg);
  count_z<<<(NATOMS + 255) / 256, 256, 0, stream>>>(z, zcnt);
  build_tables<<<1, 1, 0, stream>>>(deg, rowptr, fillpos, zcnt, zstart, zfill,
                                    tz, ts, tc, ntile, tz3, ts3, tc3, ntile3);
  fill_eidx<<<(E + 255) / 256, 256, 0, stream>>>(edst, E, fillpos, eidx);
  fill_perm<<<(NATOMS + 255) / 256, 256, 0, stream>>>(z, zfill, perm);
  build_perm3<<<(NATOMS + 255) / 256, 256, 0, stream>>>(perm, perm3);

  // ---- geometry, radial embedding, initial features
  edge_geom<<<(E + 255) / 256, 256, 0, stream>>>(pos, esrc, edst, E, eY0, eY1,
                                                 eY2, soh);
  hipMemsetAsync(h1, 0, NATOMS * 12, stream);
  h1_accum<<<(E + 255) / 256, 256, 0, stream>>>(esrc, eY1, h1, E);
  init_nodes<<<NATOMS, 256, 0, stream>>>(z, mn_s, mn_vo, h1, s_h, vo_h, ve_h);
  gemm_edge<1><<<dim3(Epad / 16, 8), 32, 0, stream>>>(soh, 64, bt_r1, 64, emb1,
                                                      128, E, 2, 0.14142136f);
  gemm_edge<0><<<dim3(Epad / 16, 8), 32, 0, stream>>>(emb1, 128, bt_r2, 128,
                                                      emb, 128, E, 4,
                                                      0.088388348f);

  // ---- three interaction layers
  for (int li = 0; li < 3; ++li) {
    lin_grouped<<<dim3(NTMAX, 18), 32, 0, stream>>>(
        s_h, 256, bt_sis[li], 288LL * 256, 256, s_si, 288, perm, tz, ts, tc,
        ntile, zrow, 8, 0.0625f);
    lin_grouped<<<dim3(NTMAX, 16), 32, 0, stream>>>(
        s_h, 256, bt_l1s[li], 256LL * 256, 256, s1, 256, perm, tz, ts, tc,
        ntile, zrow, 8, 0.0625f);
    lin_grouped<<<dim3(NT3MAX, 1), 32, 0, stream>>>(
        vo_h, 32, bt_svo[li], 16LL * 32, 32, vo_si, 16, perm3, tz3, ts3, tc3,
        ntile3, zrow, 1, 0.25f);
    lin_grouped<<<dim3(NT3MAX, 1), 32, 0, stream>>>(
        ve_h, 32, bt_sve[li], 16LL * 32, 32, ve_si, 16, perm3, tz3, ts3, tc3,
        ntile3, zrow, 1, 0.25f);
    lin_grouped<<<dim3(NT3MAX, 1), 32, 0, stream>>>(
        vo_h, 32, bt_l1vo[li], 16LL * 32, 32, vo1, 16, perm3, tz3, ts3, tc3,
        ntile3, zrow, 1, 0.25f);
    lin_grouped<<<dim3(NT3MAX, 1), 32, 0, stream>>>(
        ve_h, 32, bt_l1ve[li], 16LL * 32, 32, ve1, 16, perm3, tz3, ts3, tc3,
        ntile3, zrow, 1, 0.25f);
    gemm_edge<0><<<dim3(Epad / 16, 39), 32, 0, stream>>>(
        emb, 128, bt_tp[li], 128, wbuf, 624, E, 4, 1.0f);
    conv_aggregate<<<NATOMS, 256, 0, stream>>>(rowptr, eidx, esrc, eY0, eY1,
                                               eY2, wbuf, s1, vo1, ve1, ns_h,
                                               nvo_h, nve_h);
    lin_grouped<<<dim3(NTMAX, 18), 32, 0, stream>>>(
        ns_h, 288, bt_l2s[li], 288LL * 288, 288, l2s, 288, perm, tz, ts, tc,
        ntile, zrow, 9, INV272);
    lin_grouped<<<dim3(NT3MAX, 1), 32, 0, stream>>>(
        nvo_h, 288, bt_l2vo[li], 16LL * 288, 288, l2vo, 16, perm3, tz3, ts3,
        tc3, ntile3, zrow, 9, INV272);
    lin_grouped<<<dim3(NT3MAX, 1), 32, 0, stream>>>(
        nve_h, 32, bt_l2ve[li], 16LL * 32, 32, l2ve, 16, perm3, tz3, ts3, tc3,
        ntile3, zrow, 1, 0.25f);
    update_nodes<<<NATOMS, 288, 0, stream>>>(s_si, l2s, vo_si, l2vo, ve_si,
                                             l2ve, s_h, vo_h, ve_h);
  }

  // ---- output head
  lin_grouped<<<dim3(NTMAX, 16), 32, 0, stream>>>(
      s_h, 256, bt_fl1s, 256LL * 256, 256, s1, 256, perm, tz, ts, tc, ntile,
      zrow, 8, 0.0625f);
  lin_grouped<<<dim3(NT3MAX, 1), 32, 0, stream>>>(
      vo_h, 32, bt_fl1vo, 16LL * 32, 32, vo1, 16, perm3, tz3, ts3, tc3, ntile3,
      zrow, 1, 0.25f);
  lin_grouped<<<dim3(NT3MAX, 1), 32, 0, stream>>>(
      ve_h, 32, bt_fl1ve, 16LL * 32, 32, ve1, 16, perm3, tz3, ts3, tc3, ntile3,
      zrow, 1, 0.25f);
  gemm_edge<0><<<dim3(Epad / 16, 18), 32, 0, stream>>>(emb, 128, bt_ftp, 128,
                                                       wbuf, 288, E, 4, 1.0f);
  fin_aggregate<<<NATOMS, 288, 0, stream>>>(rowptr, eidx, esrc, eY0, eY1, wbuf,
                                            s1, vo1, ve1, s_h, z, Fsi_s, Fl2_s,
                                            Fl2_o, hnode);
  hipMemsetAsync(d_out, 0, (size_t)NGRAPH * 4, stream);
  graph_out<<<(NATOMS + 255) / 256, 256, 0, stream>>>(hnode, batch,
                                                      (float*)d_out);
}
